// QualityControlDetector_73547019976743
// MI455X (gfx1250) — matrix operand order repaired and checked by reading, not compiled, not run on hardware
//
#include <hip/hip_runtime.h>
#include <hip/hip_bf16.h>
#include <math.h>

typedef _Float16 f16;
typedef __attribute__((ext_vector_type(16))) _Float16 v16h;
typedef __attribute__((ext_vector_type(8)))  float    v8f;

#define BB   4
#define NPTS 4096
#define HH   256
#define GG   64
#define KNN  6
#define CC   5

// ---------------------------------------------------------------------------
// CDNA5 wave32 WMMA helpers. Fragments are stored "fragment-major":
// frag fi -> 32 lanes x 16 f16 contiguous (1 KB), so a fragment load is one
// aligned 32-byte v16h load (2x b128) per lane.
// ---------------------------------------------------------------------------
__device__ __forceinline__ v8f wmma16(v16h a, v16h b, v8f c) {
  return __builtin_amdgcn_wmma_f32_16x16x32_f16(
      /*neg_a=*/false, a, /*neg_b=*/false, b,
      /*c_mod=*/(short)0, c, /*reuse_a=*/false, /*reuse_b=*/false);
}

// Swizzled offset of logical B element (k, n); fragment id = ks*ntiles + nt.
// B layout (this file's contiguous k order): lane = n&15 | ((k&31)>>4)<<4 ; half j = k&15.
__device__ __forceinline__ size_t swzB_off(int k, int n, int ntiles) {
  int ks   = k >> 5, kin = k & 31;
  int lane = (n & 15) | ((kin >> 4) << 4);
  int j    = kin & 15;
  int nt   = n >> 4;
  return ((size_t)(ks * ntiles + nt) * 32 + lane) * 16 + j;
}

// Swizzled offset of logical A element (m, k); fragment id = mt*ksteps + ks.
// A layout: lane = m&15 | ((k&31)>>4)<<4 ; half j = k&15.
__device__ __forceinline__ size_t swzA_off(int m, int k, int ksteps) {
  int mt   = m >> 4, ks = k >> 5, kin = k & 31;
  int lane = (m & 15) | ((kin >> 4) << 4);
  int j    = kin & 15;
  return ((size_t)(mt * ksteps + ks) * 32 + lane) * 16 + j;
}

__device__ __forceinline__ v16h ldfrag(const f16* base, int fi, int lane) {
  return *(const v16h*)(base + ((size_t)fi * 32 + lane) * 16);
}

// ---------------------------------------------------------------------------
// Prep: swizzle f32 weight matrices into fragment-major f16 once per call.
// ---------------------------------------------------------------------------
__global__ void k_swzB(const float* __restrict__ W, int K, int N, f16* __restrict__ out) {
  int idx = blockIdx.x * blockDim.x + threadIdx.x;
  if (idx >= K * N) return;
  int k = idx / N, n = idx - k * N;
  out[swzB_off(k, n, N >> 4)] = (f16)W[idx];
}

// logical A[m][k] = W[k*M + m] (transposed storage)
__global__ void k_swzAT(const float* __restrict__ W, int M, int K, f16* __restrict__ out) {
  int idx = blockIdx.x * blockDim.x + threadIdx.x;
  if (idx >= M * K) return;
  int m = idx % M, k = idx / M;
  out[swzA_off(m, k, K >> 5)] = (f16)W[k * M + m];
}

// ---------------------------------------------------------------------------
// Kernel 0: per-batch grid argmin + winner (max point index per grid cell)
// ---------------------------------------------------------------------------
__global__ void k_winner(const float* __restrict__ pc, const float* __restrict__ grid,
                         int* __restrict__ winner) {
  __shared__ float gp[GG * 3];
  __shared__ int   win[GG];
  int b = blockIdx.x;
  int t = threadIdx.x;  // 0..63
  if (t < GG) {
    gp[t * 3 + 0] = grid[t * 3 + 0];
    gp[t * 3 + 1] = grid[t * 3 + 1];
    gp[t * 3 + 2] = grid[t * 3 + 2];
    win[t] = -1;
  }
  __syncthreads();
  const float* p = pc + (size_t)b * NPTS * 3 + t * 3;
  float x = p[0], y = p[1], z = p[2];
  float best = 3.4e38f;
  int   gi   = 0;
  for (int g = 0; g < GG; ++g) {
    float dx = x - gp[g * 3 + 0];
    float dy = y - gp[g * 3 + 1];
    float dz = z - gp[g * 3 + 2];
    float d  = dx * dx + dy * dy + dz * dz;
    if (d < best) { best = d; gi = g; }
  }
  atomicMax(&win[gi], t);
  __syncthreads();
  if (t < GG) winner[b * GG + t] = win[t];
}

// ---------------------------------------------------------------------------
// Kernel 1: kNN (K=6) over full cloud for first 64 points -> PCA normal
// ---------------------------------------------------------------------------
__device__ __forceinline__ void smallest_eigvec(float a, float bq, float cq,
                                                float d, float e, float f,
                                                float* nx, float* ny, float* nz) {
  float p1 = bq * bq + cq * cq + e * e;
  float vx, vy, vz;
  if (p1 < 1e-20f) {
    vx = (a <= d && a <= f) ? 1.f : 0.f;
    vy = (d <  a && d <= f) ? 1.f : 0.f;
    vz = (vx == 0.f && vy == 0.f) ? 1.f : 0.f;
  } else {
    float q  = (a + d + f) * (1.f / 3.f);
    float p2 = (a - q) * (a - q) + (d - q) * (d - q) + (f - q) * (f - q) + 2.f * p1;
    float p  = sqrtf(p2 * (1.f / 6.f));
    float ip = 1.f / p;
    float b00 = (a - q) * ip, b11 = (d - q) * ip, b22 = (f - q) * ip;
    float b01 = bq * ip, b02 = cq * ip, b12 = e * ip;
    float detB = b00 * (b11 * b22 - b12 * b12)
               - b01 * (b01 * b22 - b12 * b02)
               + b02 * (b01 * b12 - b11 * b02);
    float r = fminf(1.f, fmaxf(-1.f, 0.5f * detB));
    float phi = acosf(r) * (1.f / 3.f);
    float e3 = q + 2.f * p * cosf(phi + 2.0943951023931953f);  // smallest eig
    float m00 = a - e3, m11 = d - e3, m22 = f - e3;
    float c0x = bq * e  - cq * m11, c0y = cq * bq - m00 * e,   c0z = m00 * m11 - bq * bq;
    float c1x = bq * m22 - cq * e,  c1y = cq * cq - m00 * m22, c1z = m00 * e  - bq * cq;
    float c2x = m11 * m22 - e * e,  c2y = e * cq - bq * m22,   c2z = bq * e  - m11 * cq;
    float n0 = c0x * c0x + c0y * c0y + c0z * c0z;
    float n1 = c1x * c1x + c1y * c1y + c1z * c1z;
    float n2 = c2x * c2x + c2y * c2y + c2z * c2z;
    vx = c0x; vy = c0y; vz = c0z; float nb = n0;
    if (n1 > nb) { vx = c1x; vy = c1y; vz = c1z; nb = n1; }
    if (n2 > nb) { vx = c2x; vy = c2y; vz = c2z; nb = n2; }
    if (nb < 1e-30f) { vx = 0.f; vy = 0.f; vz = 1.f; }
  }
  float inv = 1.f / sqrtf(fmaxf(vx * vx + vy * vy + vz * vz, 1e-30f));
  *nx = vx * inv; *ny = vy * inv; *nz = vz * inv;
}

__global__ void k_normals(const float* __restrict__ pc, float* __restrict__ normals) {
  __shared__ float cx[64], cy[64], cz[64];
  int b = blockIdx.x;
  int t = threadIdx.x;  // 0..63, one query each
  const float* pcb = pc + (size_t)b * NPTS * 3;
  float qx = pcb[t * 3 + 0], qy = pcb[t * 3 + 1], qz = pcb[t * 3 + 2];
  float bd[KNN];
  int   bi[KNN];
#pragma unroll
  for (int k = 0; k < KNN; ++k) { bd[k] = 3.4e38f; bi[k] = 0; }
  for (int c0 = 0; c0 < NPTS; c0 += 64) {
    cx[t] = pcb[(c0 + t) * 3 + 0];
    cy[t] = pcb[(c0 + t) * 3 + 1];
    cz[t] = pcb[(c0 + t) * 3 + 2];
    __syncthreads();
    for (int j = 0; j < 64; ++j) {
      float dx = qx - cx[j], dy = qy - cy[j], dz = qz - cz[j];
      float d2 = dx * dx + dy * dy + dz * dz;
      if (d2 < bd[KNN - 1]) {
        int pos = KNN - 1;
        while (pos > 0 && bd[pos - 1] > d2) {
          bd[pos] = bd[pos - 1]; bi[pos] = bi[pos - 1]; --pos;
        }
        bd[pos] = d2; bi[pos] = c0 + j;
      }
    }
    __syncthreads();
  }
  float px[KNN], py[KNN], pz[KNN];
  float mx = 0.f, my = 0.f, mz = 0.f;
#pragma unroll
  for (int k = 0; k < KNN; ++k) {
    px[k] = pcb[bi[k] * 3 + 0];
    py[k] = pcb[bi[k] * 3 + 1];
    pz[k] = pcb[bi[k] * 3 + 2];
    mx += px[k]; my += py[k]; mz += pz[k];
  }
  const float s = 1.f / (float)KNN;
  mx *= s; my *= s; mz *= s;
  float cxx = 0.f, cxy = 0.f, cxz = 0.f, cyy = 0.f, cyz = 0.f, czz = 0.f;
#pragma unroll
  for (int k = 0; k < KNN; ++k) {
    float dx = px[k] - mx, dy = py[k] - my, dz = pz[k] - mz;
    cxx += dx * dx; cxy += dx * dy; cxz += dx * dz;
    cyy += dy * dy; cyz += dy * dz; czz += dz * dz;
  }
  float nx, ny, nz;
  smallest_eigvec(cxx, cxy, cxz, cyy, cyz, czz, &nx, &ny, &nz);
  float* o = normals + ((size_t)b * GG + t) * 3;
  o[0] = nx; o[1] = ny; o[2] = nz;
}

// ---------------------------------------------------------------------------
// Kernel 2: MLP3 (3->64->128->256) for 64 points/batch.
// Layer1 VALU -> A-frag LDS; layers 2-3 WMMA with pre-swizzled f16 weights.
// grid = B, block = 128 (4 waves, one 16-point group per wave).
// ---------------------------------------------------------------------------
__global__ void __launch_bounds__(128)
k_feat(const float* __restrict__ X, int xbstride, int mode_ones,
       const float* __restrict__ W1, const float* __restrict__ b1,
       const f16* __restrict__ W2s, const float* __restrict__ b2,
       const f16* __restrict__ W3s, const float* __restrict__ b3,
       float* __restrict__ f64out, int coloff) {
  __shared__ f16 a1s[4][2 * 512];   // 2 A-frags/wave (16x64), 8 KB total
  __shared__ f16 a2s[4][4 * 512];   // 4 A-frags/wave (16x128), 16 KB total
  int b    = blockIdx.x;
  int wave = threadIdx.x >> 5;
  int lane = threadIdx.x & 31;
  f16* a1 = a1s[wave];
  f16* a2 = a2s[wave];
  const float* xb = X + (size_t)b * xbstride;

  // Layer 1: (16x3)@(3x64)+b1, relu; write straight into A-fragment order
#pragma unroll
  for (int tt = 0; tt < 32; ++tt) {
    int flat = tt * 32 + lane;
    int row = flat >> 6, col = flat & 63;
    int n = wave * 16 + row;
    float x0, x1, x2;
    if (mode_ones) { x0 = 1.f; x1 = 1.f; x2 = 1.f; }
    else { x0 = xb[n * 3 + 0]; x1 = xb[n * 3 + 1]; x2 = xb[n * 3 + 2]; }
    float v = b1[col] + x0 * W1[col] + x1 * W1[64 + col] + x2 * W1[128 + col];
    a1[swzA_off(row, col, 2)] = (f16)fmaxf(v, 0.f);
  }
  __syncthreads();

  // Layer 2: (16x64)@(64x128)+b2, relu  (W2s frag id = ks*8 + nt)
#pragma unroll 1
  for (int nt = 0; nt < 8; ++nt) {
    v8f c = {};
#pragma unroll
    for (int ks = 0; ks < 2; ++ks) {
      c = wmma16(ldfrag(a1, ks, lane), ldfrag(W2s, ks * 8 + nt, lane), c);
    }
#pragma unroll
    for (int i = 0; i < 8; ++i) {
      int m   = i + ((lane >> 4) << 3);
      int col = nt * 16 + (lane & 15);
      a2[swzA_off(m, col, 4)] = (f16)fmaxf(c[i] + b2[col], 0.f);
    }
  }
  __syncthreads();

  // Layer 3: (16x128)@(128x256)+b3 (no relu)  (W3s frag id = ks*16 + nt)
#pragma unroll 1
  for (int nt = 0; nt < 16; ++nt) {
    v8f c = {};
#pragma unroll
    for (int ks = 0; ks < 4; ++ks) {
      c = wmma16(ldfrag(a2, ks, lane), ldfrag(W3s, ks * 16 + nt, lane), c);
    }
#pragma unroll
    for (int i = 0; i < 8; ++i) {
      int m   = i + ((lane >> 4) << 3);
      int col = nt * 16 + (lane & 15);
      int n   = wave * 16 + m;
      f64out[((size_t)b * GG + n) * 768 + coloff + col] = c[i] + b3[col];
    }
  }
}

// ---------------------------------------------------------------------------
// Kernel 3: gather winners -> gf_t output (f32) + swizzled f16 B staging
// ---------------------------------------------------------------------------
__global__ void k_gather(const float* __restrict__ f64, const int* __restrict__ winner,
                         float* __restrict__ gft_out, f16* __restrict__ gfBs) {
  int idx = blockIdx.x * blockDim.x + threadIdx.x;
  int total = BB * 768 * GG;
  if (idx >= total) return;
  int g = idx & 63;
  int c = (idx >> 6) % 768;
  int b = idx / (768 * GG);
  int wn = winner[b * GG + g];
  float v = 0.f;
  if (wn >= 0) v = f64[((size_t)b * GG + wn) * 768 + c];
  gft_out[((size_t)b * 768 + c) * GG + g] = v;
  gfBs[(size_t)b * 768 * GG + swzB_off(c, g, 4)] = (f16)v;  // B(768x64), ntiles=4
}

// ---------------------------------------------------------------------------
// Kernel 4: dense trunk. h = relu(W1^T(256x768)@gf(768x64)+b1)
//           processed = W2^T(256x256)@h+b2. All operands pre-swizzled f16.
// grid = B, block = 256 (8 waves)
// ---------------------------------------------------------------------------
__global__ void __launch_bounds__(256)
k_dense(const f16* __restrict__ gfBs,
        const f16* __restrict__ W1s, const float* __restrict__ b1,
        const f16* __restrict__ W2s, const float* __restrict__ b2,
        float* __restrict__ proc_out, f16* __restrict__ proc16s) {
  __shared__ f16 hbuf[32 * 512];  // h as 32 B-frags (256x64), 32 KB
  int b    = blockIdx.x;
  int wave = threadIdx.x >> 5;
  int lane = threadIdx.x & 31;
  const f16* gb = gfBs + (size_t)b * 768 * GG;

  // Phase 1: 64 tiles (mt 0..15, nt 0..3), K=768 -> 24 wmma steps
#pragma unroll 1
  for (int it = 0; it < 8; ++it) {
    int tile = it * 8 + wave;
    int mt = tile >> 2, nt = tile & 3;
    v8f c = {};
#pragma unroll 4
    for (int ks = 0; ks < 24; ++ks) {
      c = wmma16(ldfrag(W1s, mt * 24 + ks, lane),
                 ldfrag(gb, ks * 4 + nt, lane), c);
    }
#pragma unroll
    for (int i = 0; i < 8; ++i) {
      int m = mt * 16 + i + ((lane >> 4) << 3);
      int g = nt * 16 + (lane & 15);
      hbuf[swzB_off(m, g, 4)] = (f16)fmaxf(c[i] + b1[m], 0.f);  // bias by h-row
    }
  }
  __syncthreads();

  // Phase 2: processed = W2^T @ h + b2, K=256 -> 8 wmma steps
#pragma unroll 1
  for (int it = 0; it < 8; ++it) {
    int tile = it * 8 + wave;
    int mt = tile >> 2, nt = tile & 3;
    v8f c = {};
#pragma unroll
    for (int ks = 0; ks < 8; ++ks) {
      c = wmma16(ldfrag(W2s, mt * 8 + ks, lane),
                 ldfrag(hbuf, ks * 4 + nt, lane), c);
    }
#pragma unroll
    for (int i = 0; i < 8; ++i) {
      int m = mt * 16 + i + ((lane >> 4) << 3);
      int g = nt * 16 + (lane & 15);
      float v = c[i] + b2[m];
      proc_out[((size_t)b * HH + m) * GG + g] = v;              // `processed`
      // df A-staging: logical A[m=g][k=m], M=64, ksteps=8
      proc16s[(size_t)b * HH * GG + swzA_off(g, m, 8)] = (f16)v;
    }
  }
}

// ---------------------------------------------------------------------------
// Kernel 5: heads. df(64x256)@W(256x128) via WMMA, then 128->5 softmax /
// 128->1 sigmoid in VALU. grid = B, block = 256 (waves 0-3: cl, 4-7: sv)
// ---------------------------------------------------------------------------
__global__ void __launch_bounds__(256)
k_heads(const f16* __restrict__ proc16s,
        const f16* __restrict__ clW1s, const float* __restrict__ clb1,
        const float* __restrict__ clW2, const float* __restrict__ clb2,
        const f16* __restrict__ svW1s, const float* __restrict__ svb1,
        const float* __restrict__ svW2, const float* __restrict__ svb2,
        float* __restrict__ probs, float* __restrict__ sev) {
  __shared__ f16 t1[GG * 128];  // linear (VALU-consumed), 16 KB
  __shared__ f16 s1[GG * 128];  // 16 KB
  int b    = blockIdx.x;
  int wave = threadIdx.x >> 5;
  int lane = threadIdx.x & 31;
  const f16* df = proc16s + (size_t)b * HH * GG;  // A-frags: fi = mt*8 + ks

  int hsel = wave >> 2;  // 0 = cl, 1 = sv
  int w4   = wave & 3;
  const f16*   Wp = hsel ? svW1s : clW1s;
  const float* bp = hsel ? svb1 : clb1;
  f16* obuf = hsel ? s1 : t1;

#pragma unroll 1
  for (int it = 0; it < 8; ++it) {
    int tile = it * 4 + w4;     // 32 tiles (mt 0..3, nt 0..7)
    int mt = tile >> 3, nt = tile & 7;
    v8f c = {};
#pragma unroll
    for (int ks = 0; ks < 8; ++ks) {
      c = wmma16(ldfrag(df, mt * 8 + ks, lane),
                 ldfrag(Wp, ks * 8 + nt, lane), c);
    }
#pragma unroll
    for (int i = 0; i < 8; ++i) {
      int m   = mt * 16 + i + ((lane >> 4) << 3);
      int col = nt * 16 + (lane & 15);
      obuf[m * 128 + col] = (f16)fmaxf(c[i] + bp[col], 0.f);
    }
  }
  __syncthreads();

  int t = threadIdx.x;
  if (t < GG) {
    float l[CC];
#pragma unroll
    for (int c = 0; c < CC; ++c) l[c] = clb2[c];
    for (int k = 0; k < 128; ++k) {
      float v = (float)t1[t * 128 + k];
#pragma unroll
      for (int c = 0; c < CC; ++c) l[c] += v * clW2[k * CC + c];
    }
    float mx = l[0];
#pragma unroll
    for (int c = 1; c < CC; ++c) mx = fmaxf(mx, l[c]);
    float ssum = 0.f;
#pragma unroll
    for (int c = 0; c < CC; ++c) { l[c] = expf(l[c] - mx); ssum += l[c]; }
    float inv = 1.f / ssum;
#pragma unroll
    for (int c = 0; c < CC; ++c) probs[((size_t)b * GG + t) * CC + c] = l[c] * inv;
  } else if (t < 2 * GG) {
    int r = t - GG;
    float v = svb2[0];
    for (int k = 0; k < 128; ++k) v += (float)s1[r * 128 + k] * svW2[k];
    sev[b * GG + r] = 1.f / (1.f + expf(-v));
  }
}

// ---------------------------------------------------------------------------
extern "C" void kernel_launch(void* const* d_in, const int* in_sizes, int n_in,
                              void* d_out, int out_size, void* d_ws, size_t ws_size,
                              hipStream_t stream) {
  const float* pc   = (const float*)d_in[0];
  const float* grid = (const float*)d_in[1];
  const float* pe[6]; const float* ne[6]; const float* te[6];
  for (int i = 0; i < 6; ++i) {
    pe[i] = (const float*)d_in[2 + i];
    ne[i] = (const float*)d_in[8 + i];
    te[i] = (const float*)d_in[14 + i];
  }
  const float* dnW1 = (const float*)d_in[20];
  const float* dnb1 = (const float*)d_in[21];
  const float* dnW2 = (const float*)d_in[22];
  const float* dnb2 = (const float*)d_in[23];
  const float* clW1 = (const float*)d_in[24];
  const float* clb1 = (const float*)d_in[25];
  const float* clW2 = (const float*)d_in[26];
  const float* clb2 = (const float*)d_in[27];
  const float* svW1 = (const float*)d_in[28];
  const float* svb1 = (const float*)d_in[29];
  const float* svW2 = (const float*)d_in[30];
  const float* svb2 = (const float*)d_in[31];

  // ---- workspace layout (all blocks 1 KB aligned) ----
  char*  ws      = (char*)d_ws;
  size_t off     = 0;
  int*   winner  = (int*)(ws + off);   off += 1024;                  // 4*64 ints
  float* normals = (float*)(ws + off); off += 4096;                  // 4*64*3 f32 (padded)
  float* f64f    = (float*)(ws + off); off += (size_t)BB * GG * 768 * 4;   // 768 KB
  f16*   gfBs    = (f16*)(ws + off);   off += (size_t)BB * 768 * GG * 2;   // 384 KB
  f16*   proc16s = (f16*)(ws + off);   off += (size_t)BB * HH * GG * 2;    // 128 KB
  // swizzled f16 weights
  f16* peW2s = (f16*)(ws + off); off += 64 * 128 * 2;
  f16* peW3s = (f16*)(ws + off); off += 128 * 256 * 2;
  f16* neW2s = (f16*)(ws + off); off += 64 * 128 * 2;
  f16* neW3s = (f16*)(ws + off); off += 128 * 256 * 2;
  f16* teW2s = (f16*)(ws + off); off += 64 * 128 * 2;
  f16* teW3s = (f16*)(ws + off); off += 128 * 256 * 2;
  f16* clW1s = (f16*)(ws + off); off += 256 * 128 * 2;
  f16* svW1s = (f16*)(ws + off); off += 256 * 128 * 2;
  f16* dnW1s = (f16*)(ws + off); off += 768 * 256 * 2;
  f16* dnW2s = (f16*)(ws + off); off += 256 * 256 * 2;

  // output layout: probs(4,64,5) | sev(4,64) | processed(4,256,64) | gf_t(4,768,64)
  float* out      = (float*)d_out;
  float* probs    = out;                    // 1280
  float* sev      = out + 1280;             // 256
  float* proc_out = out + 1536;             // 65536
  float* gft_out  = out + 67072;            // 196608

  // ---- weight swizzle prep ----
  #define SWZB(W, K, N, O) k_swzB<<<((K)*(N) + 255) / 256, 256, 0, stream>>>(W, K, N, O)
  SWZB(pe[2], 64, 128, peW2s);  SWZB(pe[4], 128, 256, peW3s);
  SWZB(ne[2], 64, 128, neW2s);  SWZB(ne[4], 128, 256, neW3s);
  SWZB(te[2], 64, 128, teW2s);  SWZB(te[4], 128, 256, teW3s);
  SWZB(clW1, 256, 128, clW1s);  SWZB(svW1, 256, 128, svW1s);
  #undef SWZB
  k_swzAT<<<(256 * 768 + 255) / 256, 256, 0, stream>>>(dnW1, 256, 768, dnW1s);
  k_swzAT<<<(256 * 256 + 255) / 256, 256, 0, stream>>>(dnW2, 256, 256, dnW2s);

  // ---- pipeline ----
  k_winner <<<BB, 64, 0, stream>>>(pc, grid, winner);
  k_normals<<<BB, 64, 0, stream>>>(pc, normals);
  k_feat<<<BB, 128, 0, stream>>>(pc, NPTS * 3, 0,
      pe[0], pe[1], peW2s, pe[3], peW3s, pe[5], f64f, 0);
  k_feat<<<BB, 128, 0, stream>>>(normals, GG * 3, 0,
      ne[0], ne[1], neW2s, ne[3], neW3s, ne[5], f64f, 256);
  k_feat<<<BB, 128, 0, stream>>>(pc, 0, 1,
      te[0], te[1], teW2s, te[3], teW3s, te[5], f64f, 512);
  int total = BB * 768 * GG;
  k_gather<<<(total + 255) / 256, 256, 0, stream>>>(f64f, winner, gft_out, gfBs);
  k_dense<<<BB, 256, 0, stream>>>(gfBs, dnW1s, dnb1, dnW2s, dnb2, proc_out, proc16s);
  k_heads<<<BB, 256, 0, stream>>>(proc16s, clW1s, clb1, clW2, clb2,
                                  svW1s, svb1, svW2, svb2, probs, sev);
}
